// ContinuousThoughtMachine_51651276701812
// MI455X (gfx1250) — compile-verified, matrix operand order checked by reading, not run on hardware
//
#include <hip/hip_runtime.h>

// ---------------------------------------------------------------------------
// Continuous Thought Machine for MI455X (gfx1250, wave32, WMMA).
// All dense projections are f16 WMMA GEMMs (D = X * W^T, f32 accumulate);
// weights converted once per call to transposed f16 with padded strides so
// both operands load as contiguous b128 pairs matching the wave32 fragment
// layout. GEMM waves are register-blocked along N (NB tiles/wave) to reuse
// the A fragment; per K-step all loads are issued before the WMMA group so
// the scheduler can overlap global_load_b128 with v_wmma via partial waits.
// Weights (~24 MB f16) stay resident in the 192 MB L2 across the 32 ticks.
// ---------------------------------------------------------------------------

typedef __attribute__((ext_vector_type(16))) _Float16 v16h;
typedef __attribute__((ext_vector_type(8)))  _Float16 v8h;
typedef __attribute__((ext_vector_type(8)))  float    v8f;

namespace ctm {
constexpr int B = 128, S = 128, DB = 512, D = 512, H = 8, DH = 64;
constexpr int N = 2048, M = 32, T = 32;
constexpr int KO = 1024, KA = 1024, C = 1000;
constexpr int W0 = 2048, W1 = 1032, W2 = 16;
constexpr int CAT = D + N;     // 2560
constexpr int W1P = 1056;      // W1 padded (mult of 32 for K, of 16 for N)
constexpr int W2P = 32;        // W2 padded to one K-step
constexpr int CP  = 1008;      // C padded to 63 N-tiles
constexpr int BS  = B * S;
}
using namespace ctm;

// f32 -> f16 elementwise
__global__ void k_tohalf(const float* __restrict__ x, _Float16* __restrict__ y, int n) {
  int i = blockIdx.x * 256 + threadIdx.x;
  if (i < n) y[i] = (_Float16)x[i];
}

// W (K x Nout, f32, row-major) -> Wt (NoutPad x Kpad, f16), zero-padded.
__global__ void k_wtrans(const float* __restrict__ W, _Float16* __restrict__ Wt,
                         int K, int Nout, int Kpad, int NoutPad) {
  int i = blockIdx.x * 256 + threadIdx.x;
  if (i >= NoutPad * Kpad) return;
  int n = i / Kpad, k = i - n * Kpad;
  _Float16 v = (_Float16)0.f;
  if (n < Nout && k < K) v = (_Float16)W[(long)k * Nout + n];
  Wt[i] = v;
}

// broadcast init_hist (N,M) across batch into ring (B,N,M)
__global__ void k_ring_init(const float* __restrict__ ih, float* __restrict__ ring) {
  int i = blockIdx.x * 256 + threadIdx.x;
  if (i >= B * N * M) return;
  ring[i] = ih[i % (N * M)];
}

// One wave per (1 m-tile x NB n-tiles): D = X(M x K) * Wt(N x K)^T + bias.
// A-fragment (16-bit, wave32): lane&15 = row, halves 0..7 -> K = hi*8+0..7,
// halves 8..15 -> K = hi*8+16..23 (hi = lane>>4). B-fragment symmetric with
// lane&15 = output column (Wt row). One A fragment feeds NB WMMAs per K-step;
// all loads of a K-step are issued before its WMMA group (latency overlap).
template <int NB>
__global__ void k_wmma_gemm(const _Float16* __restrict__ X, int ldx,
                            const _Float16* __restrict__ Wt, int ldw,
                            const float* __restrict__ bias,
                            float* __restrict__ Yf, int ldyf,
                            _Float16* __restrict__ Yh, int ldyh,
                            int Nvalid, int Ksteps) {
  const int lane = threadIdx.x;
  const int r = lane & 15, hi = lane >> 4;
  const int mtile = blockIdx.y;
  const int n0 = blockIdx.x * NB;
  const long wrow = (long)16 * ldw;                       // n-tile stride in Wt
  const _Float16* pa = X  + (long)(mtile * 16 + r) * ldx + hi * 8;
  const _Float16* pw = Wt + ((long)n0 * 16 + r) * ldw + hi * 8;
#pragma unroll
  for (int j = 0; j < NB; ++j)
    __builtin_prefetch(pw + j * wrow, 0, 1);   // global_prefetch_b8: warm tiles
  v8f acc[NB] = {};
  for (int k = 0; k < Ksteps; ++k) {
    const long ko = (long)k * 32;
    // issue all fragment loads for this K-step first
    v8h a0 = *(const v8h*)(pa + ko);
    v8h a1 = *(const v8h*)(pa + ko + 16);
    v8h b0[NB], b1[NB];
#pragma unroll
    for (int j = 0; j < NB; ++j) {
      b0[j] = *(const v8h*)(pw + j * wrow + ko);
      b1[j] = *(const v8h*)(pw + j * wrow + ko + 16);
    }
    v16h A = __builtin_shufflevector(a0, a1, 0,1,2,3,4,5,6,7,8,9,10,11,12,13,14,15);
#pragma unroll
    for (int j = 0; j < NB; ++j) {
      v16h Bm = __builtin_shufflevector(b0[j], b1[j],
                                        0,1,2,3,4,5,6,7,8,9,10,11,12,13,14,15);
      acc[j] = __builtin_amdgcn_wmma_f32_16x16x32_f16(false, A, false, Bm,
                                                      (short)0, acc[j], false, false);
    }
  }
#pragma unroll
  for (int j = 0; j < NB; ++j) {
    const int col = (n0 + j) * 16 + r;
    if (col >= Nvalid) continue;       // pads stay zero (workspace memset)
    const float bv = bias ? bias[col] : 0.f;
#pragma unroll
    for (int v = 0; v < 8; ++v) {      // C layout: lane&15 = col, VGPR v + hi*8 = row
      int row = mtile * 16 + hi * 8 + v;
      float y = acc[j][v] + bv;
      if (Yf) Yf[(long)row * ldyf + col] = y;
      if (Yh) Yh[(long)row * ldyh + col] = (_Float16)y;
    }
  }
}

// Row LayerNorm (+optional residual pre-add, +optional SiLU); f32 and/or f16
// outputs; csf = column stride for the f32 store (used for ring writes).
__global__ void k_ln(const float* __restrict__ X, int ldx,
                     const float* __restrict__ R, int ldr,
                     const float* __restrict__ g, const float* __restrict__ be,
                     int L, int dosilu,
                     float* __restrict__ Yf, int ldyf, int csf,
                     _Float16* __restrict__ Yh, int ldyh) {
  const int row = blockIdx.x, tid = threadIdx.x;
  __shared__ float red[256];
  const float* xr = X + (long)row * ldx;
  const float* rr = R ? R + (long)row * ldr : nullptr;
  float s = 0.f;
  for (int c = tid; c < L; c += 256) { float x = xr[c]; if (rr) x += rr[c]; s += x; }
  red[tid] = s; __syncthreads();
  for (int o = 128; o > 0; o >>= 1) { if (tid < o) red[tid] += red[tid + o]; __syncthreads(); }
  const float mu = red[0] / L;
  __syncthreads();
  float vs = 0.f;
  for (int c = tid; c < L; c += 256) { float x = xr[c]; if (rr) x += rr[c]; x -= mu; vs += x * x; }
  red[tid] = vs; __syncthreads();
  for (int o = 128; o > 0; o >>= 1) { if (tid < o) red[tid] += red[tid + o]; __syncthreads(); }
  const float rstd = rsqrtf(red[0] / L + 1e-5f);
  for (int c = tid; c < L; c += 256) {
    float x = xr[c]; if (rr) x += rr[c];
    float y = (x - mu) * rstd * g[c];
    if (be) y += be[c];
    if (dosilu) y = y / (1.f + __expf(-y));
    if (Yf) Yf[(long)row * ldyf + (long)c * csf] = y;
    if (Yh) Yh[(long)row * ldyh + c] = (_Float16)y;
  }
}

// Per-neuron GLU over sliding history (ring buffer, logical start = slot0).
// Writes zp (f32) and the f16 copy straight into the concat buffer.
__global__ void k_nlm(const float* __restrict__ ring, int slot0,
                      const float* __restrict__ w1, const float* __restrict__ b1,
                      const float* __restrict__ tempp,
                      float* __restrict__ zp, _Float16* __restrict__ zph, int ldzh) {
  int i = blockIdx.x * 256 + threadIdx.x;
  if (i >= B * N) return;
  int b = i >> 11, n = i & (N - 1);
  const float* h = ring + (long)i * M;
  float o0 = b1[n * 2 + 0], o1 = b1[n * 2 + 1];
#pragma unroll
  for (int m = 0; m < M; ++m) {
    float x = h[(slot0 + m) & (M - 1)];
    o0 += x * w1[(long)(m * 2 + 0) * N + n];
    o1 += x * w1[(long)(m * 2 + 1) * N + n];
  }
  float z = o0 * (1.f / (1.f + __expf(-o1))) / tempp[0];
  zp[i] = z;
  zph[(long)b * ldzh + n] = (_Float16)z;
}

// sync_update: EMA of paired products; in-place a/bsum; f16 normalized out.
__global__ void k_sync(const float* __restrict__ zp,
                       const int* __restrict__ li, const int* __restrict__ ri,
                       const float* __restrict__ decay,
                       float* __restrict__ a, float* __restrict__ bsum,
                       _Float16* __restrict__ outh, int K) {
  int i = blockIdx.x * 256 + threadIdx.x;
  if (i >= B * K) return;
  int b = i / K, k = i - b * K;
  const float* z = zp + (long)b * N;
  float p = z[li[k]] * z[ri[k]];
  float d = fminf(fmaxf(decay[k], 0.f), 15.f);
  float r = __expf(-d);
  float av = a[i] * r + p;
  float bv = bsum[i] * r + 1.f;
  a[i] = av; bsum[i] = bv;
  outh[i] = (_Float16)(av * rsqrtf(bv));
}

// Single-query attention per (b,h): softmax(q.K/8).V  -> ctx f16 (B,D)
__global__ void k_attn(const float* __restrict__ qh,
                       const _Float16* __restrict__ Kh,
                       const _Float16* __restrict__ Vh,
                       _Float16* __restrict__ ctx) {
  const int b = blockIdx.x >> 3, hh = blockIdx.x & 7;
  const int tid = threadIdx.x;  // 128
  __shared__ float qs[DH];
  __shared__ float sc[S];
  __shared__ float red[128];
  if (tid < DH) qs[tid] = qh[(long)b * D + hh * DH + tid];
  __syncthreads();
  {
    const _Float16* kp = Kh + ((long)b * S + tid) * D + hh * DH;
    float acc = 0.f;
#pragma unroll 8
    for (int d = 0; d < DH; ++d) acc += qs[d] * (float)kp[d];
    sc[tid] = acc * 0.125f;  // 1/sqrt(64)
  }
  red[tid] = sc[tid]; __syncthreads();
  for (int o = 64; o > 0; o >>= 1) { if (tid < o) red[tid] = fmaxf(red[tid], red[tid + o]); __syncthreads(); }
  float mx = red[0]; __syncthreads();
  float e = __expf(sc[tid] - mx);
  sc[tid] = e; red[tid] = e; __syncthreads();
  for (int o = 64; o > 0; o >>= 1) { if (tid < o) red[tid] += red[tid + o]; __syncthreads(); }
  float inv = 1.f / red[0];
  __syncthreads();
  if (tid < DH) {
    const _Float16* vp = Vh + (long)b * S * D + hh * DH + tid;
    float acc = 0.f;
    for (int s = 0; s < S; ++s) acc += sc[s] * (float)vp[(long)s * D];
    ctx[(long)b * D + hh * DH + tid] = (_Float16)(acc * inv);
  }
}

extern "C" void kernel_launch(void* const* d_in, const int* in_sizes, int n_in,
                              void* d_out, int out_size, void* d_ws, size_t ws_size,
                              hipStream_t stream) {
  (void)in_sizes; (void)n_in; (void)out_size;
  const float* features   = (const float*)d_in[0];
  const float* q_w        = (const float*)d_in[1];
  const float* q_b        = (const float*)d_in[2];
  const float* kv_w       = (const float*)d_in[3];
  const float* kv_b       = (const float*)d_in[4];
  const float* kv_g       = (const float*)d_in[5];
  const float* kv_beta    = (const float*)d_in[6];
  const float* attn_q_w   = (const float*)d_in[7];
  const float* attn_q_b   = (const float*)d_in[8];
  const float* attn_k_w   = (const float*)d_in[9];
  const float* attn_k_b   = (const float*)d_in[10];
  const float* attn_v_w   = (const float*)d_in[11];
  const float* attn_v_b   = (const float*)d_in[12];
  const float* attn_o_w   = (const float*)d_in[13];
  const float* attn_o_b   = (const float*)d_in[14];
  const float* init_hist  = (const float*)d_in[15];
  const float* nlm_w1     = (const float*)d_in[16];
  const float* nlm_b1     = (const float*)d_in[17];
  const float* nlm_temp   = (const float*)d_in[18];
  const float* syn_in_w   = (const float*)d_in[19];
  const float* syn_in_g   = (const float*)d_in[20];
  const float* down0_w    = (const float*)d_in[21];
  const float* down0_b    = (const float*)d_in[22];
  const float* down0_g    = (const float*)d_in[23];
  const float* down0_beta = (const float*)d_in[24];
  const float* down1_w    = (const float*)d_in[25];
  const float* down1_b    = (const float*)d_in[26];
  const float* down1_g    = (const float*)d_in[27];
  const float* down1_beta = (const float*)d_in[28];
  const float* up0_w      = (const float*)d_in[29];
  const float* up0_b      = (const float*)d_in[30];
  const float* up0_g      = (const float*)d_in[31];
  const float* up0_beta   = (const float*)d_in[32];
  const float* up1_w      = (const float*)d_in[33];
  const float* up1_b      = (const float*)d_in[34];
  const float* up1_g      = (const float*)d_in[35];
  const float* up1_beta   = (const float*)d_in[36];
  const float* skip0_g    = (const float*)d_in[37];
  const float* skip0_b    = (const float*)d_in[38];
  const float* skip1_g    = (const float*)d_in[39];
  const float* skip1_b    = (const float*)d_in[40];
  const float* decay_out  = (const float*)d_in[41];
  const float* decay_act  = (const float*)d_in[42];
  const float* cls_w      = (const float*)d_in[43];
  const float* cls_b      = (const float*)d_in[44];
  const int*   out_li     = (const int*)d_in[45];
  const int*   out_ri     = (const int*)d_in[46];
  const int*   act_li     = (const int*)d_in[47];
  const int*   act_ri     = (const int*)d_in[48];
  float* outp = (float*)d_out;

  // ---- workspace layout ----
  char* base = (char*)d_ws;
  size_t off = 0;
  auto alloc = [&](size_t bytes) -> void* {
    void* p = base + off; off = (off + bytes + 255) & ~(size_t)255; return p;
  };
  _Float16* feat16 = (_Float16*)alloc((size_t)BS * DB * 2);
  _Float16* kvT  = (_Float16*)alloc((size_t)512 * 512 * 2);
  _Float16* kT   = (_Float16*)alloc((size_t)512 * 512 * 2);
  _Float16* vT   = (_Float16*)alloc((size_t)512 * 512 * 2);
  _Float16* aqT  = (_Float16*)alloc((size_t)512 * 512 * 2);
  _Float16* oT   = (_Float16*)alloc((size_t)512 * 512 * 2);
  _Float16* qwT  = (_Float16*)alloc((size_t)512 * 1024 * 2);
  _Float16* synT = (_Float16*)alloc((size_t)W0 * CAT * 2);
  _Float16* d0T  = (_Float16*)alloc((size_t)W1P * W0 * 2);
  _Float16* d1T  = (_Float16*)alloc((size_t)W2 * W1P * 2);
  _Float16* u1T  = (_Float16*)alloc((size_t)W1P * W2P * 2);
  _Float16* u0T  = (_Float16*)alloc((size_t)W0 * W1P * 2);
  _Float16* clsT = (_Float16*)alloc((size_t)CP * KO * 2);
  float*    kvraw = (float*)alloc((size_t)BS * D * 4);
  _Float16* kv16  = (_Float16*)alloc((size_t)BS * D * 2);
  _Float16* Kh16  = (_Float16*)alloc((size_t)BS * D * 2);
  _Float16* Vh16  = (_Float16*)alloc((size_t)BS * D * 2);
  float*    ring  = (float*)alloc((size_t)B * N * M * 4);
  float*    zp    = (float*)alloc((size_t)B * N * 4);
  _Float16* catbuf = (_Float16*)alloc((size_t)B * CAT * 2);
  float*    aa = (float*)alloc((size_t)B * KA * 4);
  float*    ba = (float*)alloc((size_t)B * KA * 4);
  float*    ao = (float*)alloc((size_t)B * KO * 4);
  float*    bo = (float*)alloc((size_t)B * KO * 4);
  _Float16* syncA16 = (_Float16*)alloc((size_t)B * KA * 2);
  _Float16* syncO16 = (_Float16*)alloc((size_t)B * KO * 2);
  _Float16* q16   = (_Float16*)alloc((size_t)B * D * 2);
  float*    qhraw = (float*)alloc((size_t)B * D * 4);
  _Float16* ctx16 = (_Float16*)alloc((size_t)B * D * 2);
  float*    x0raw = (float*)alloc((size_t)B * W0 * 4);
  float*    x0f   = (float*)alloc((size_t)B * W0 * 4);
  _Float16* x016  = (_Float16*)alloc((size_t)B * W0 * 2);
  float*    d0raw = (float*)alloc((size_t)B * W1P * 4);
  float*    d0f   = (float*)alloc((size_t)B * W1P * 4);
  _Float16* d016  = (_Float16*)alloc((size_t)B * W1P * 2);
  float*    d1raw = (float*)alloc((size_t)B * W2P * 4);
  _Float16* d116  = (_Float16*)alloc((size_t)B * W2P * 2);
  float*    uraw  = (float*)alloc((size_t)B * W1P * 4);
  float*    uf    = (float*)alloc((size_t)B * W1P * 4);
  _Float16* u216  = (_Float16*)alloc((size_t)B * W1P * 2);
  float*    u3raw = (float*)alloc((size_t)B * W0 * 4);
  float*    u3f   = (float*)alloc((size_t)B * W0 * 4);
  size_t used = off;

  auto cdiv = [](int a, int b) { return (a + b - 1) / b; };

  // Zero workspace: clears pad columns (so they contribute 0 to WMMA K-loops)
  // and the sync EMA accumulators. Capturable (async memset).
  hipMemsetAsync(d_ws, 0, used < ws_size ? used : ws_size, stream);

  // GEMM dispatch: pick largest NB dividing the tile count (A-fragment reuse).
  auto gemm = [&](int Ntiles, int Mtiles,
                  const _Float16* X, int ldx, const _Float16* Wt, int ldw,
                  const float* bias, float* Yf, int ldyf, _Float16* Yh, int ldyh,
                  int Nvalid, int Ksteps) {
    if ((Ntiles & 3) == 0)
      k_wmma_gemm<4><<<dim3(Ntiles / 4, Mtiles), 32, 0, stream>>>(
          X, ldx, Wt, ldw, bias, Yf, ldyf, Yh, ldyh, Nvalid, Ksteps);
    else if (Ntiles % 3 == 0)
      k_wmma_gemm<3><<<dim3(Ntiles / 3, Mtiles), 32, 0, stream>>>(
          X, ldx, Wt, ldw, bias, Yf, ldyf, Yh, ldyh, Nvalid, Ksteps);
    else if ((Ntiles & 1) == 0)
      k_wmma_gemm<2><<<dim3(Ntiles / 2, Mtiles), 32, 0, stream>>>(
          X, ldx, Wt, ldw, bias, Yf, ldyf, Yh, ldyh, Nvalid, Ksteps);
    else
      k_wmma_gemm<1><<<dim3(Ntiles, Mtiles), 32, 0, stream>>>(
          X, ldx, Wt, ldw, bias, Yf, ldyf, Yh, ldyh, Nvalid, Ksteps);
  };

  // ---- weight conversion: f32 (K x N) -> f16 transposed (Npad x Kpad) ----
  auto wt = [&](const float* W, _Float16* Wt, int K, int Nout, int Kpad, int NoutPad) {
    int cnt = NoutPad * Kpad;
    k_wtrans<<<cdiv(cnt, 256), 256, 0, stream>>>(W, Wt, K, Nout, Kpad, NoutPad);
  };
  wt(kv_w,     kvT,  512, 512, 512, 512);
  wt(attn_k_w, kT,   512, 512, 512, 512);
  wt(attn_v_w, vT,   512, 512, 512, 512);
  wt(attn_q_w, aqT,  512, 512, 512, 512);
  wt(attn_o_w, oT,   512, 512, 512, 512);
  wt(q_w,      qwT,  1024, 512, 1024, 512);
  wt(syn_in_w, synT, CAT, W0, CAT, W0);
  wt(down0_w,  d0T,  W0, W1, W0, W1P);
  wt(down1_w,  d1T,  W1, W2, W1P, W2);
  wt(up1_w,    u1T,  W2, W1, W2P, W1P);
  wt(up0_w,    u0T,  W1, W0, W1P, W0);
  wt(cls_w,    clsT, KO, C, KO, CP);

  // ---- tick-invariant preamble ----
  k_tohalf<<<cdiv(BS * DB, 256), 256, 0, stream>>>(features, feat16, BS * DB);
  // kv = LN(features @ kv_w + kv_b)
  gemm(32, BS / 16, feat16, DB, kvT, DB, kv_b, kvraw, D, nullptr, 0, D, DB / 32);
  k_ln<<<BS, 256, 0, stream>>>(kvraw, D, nullptr, 0, kv_g, kv_beta, D, 0,
                               nullptr, 0, 1, kv16, D);
  gemm(32, BS / 16, kv16, D, kT, D, attn_k_b, nullptr, 0, Kh16, D, D, D / 32);
  gemm(32, BS / 16, kv16, D, vT, D, attn_v_b, nullptr, 0, Vh16, D, D, D / 32);
  k_ring_init<<<cdiv(B * N * M, 256), 256, 0, stream>>>(init_hist, ring);
  // z0 = nlm(hist0); f16 copy goes into concat buffer cols [D, D+N)
  k_nlm<<<cdiv(B * N, 256), 256, 0, stream>>>(ring, 0, nlm_w1, nlm_b1, nlm_temp,
                                              zp, catbuf + D, CAT);

  // ---- tick loop ----
  for (int t = 0; t < T; ++t) {
    // action sync -> query
    k_sync<<<cdiv(B * KA, 256), 256, 0, stream>>>(zp, act_li, act_ri, decay_act,
                                                  aa, ba, syncA16, KA);
    gemm(32, 8, syncA16, KA, qwT, KA, q_b, nullptr, 0, q16, D, D, KA / 32);
    gemm(32, 8, q16, D, aqT, D, attn_q_b, qhraw, D, nullptr, 0, D, D / 32);
    k_attn<<<B * H, 128, 0, stream>>>(qhraw, Kh16, Vh16, ctx16);
    // attended -> concat buffer cols [0, D)
    gemm(32, 8, ctx16, D, oT, D, attn_o_b, nullptr, 0, catbuf, CAT, D, D / 32);
    // synapse UNet
    gemm(W0 / 16, 8, catbuf, CAT, synT, CAT, nullptr, x0raw, W0, nullptr, 0, W0, CAT / 32);
    k_ln<<<B, 256, 0, stream>>>(x0raw, W0, nullptr, 0, syn_in_g, nullptr, W0, 1,
                                x0f, W0, 1, x016, W0);
    gemm(W1P / 16, 8, x016, W0, d0T, W0, down0_b, d0raw, W1P, nullptr, 0, W1, W0 / 32);
    k_ln<<<B, 256, 0, stream>>>(d0raw, W1P, nullptr, 0, down0_g, down0_beta, W1, 1,
                                d0f, W1P, 1, d016, W1P);
    gemm(1, 8, d016, W1P, d1T, W1P, down1_b, d1raw, W2P, nullptr, 0, W2, W1P / 32);
    k_ln<<<B, 256, 0, stream>>>(d1raw, W2P, nullptr, 0, down1_g, down1_beta, W2, 1,
                                nullptr, 0, 1, d116, W2P);
    gemm(W1P / 16, 8, d116, W2P, u1T, W2P, up1_b, uraw, W1P, nullptr, 0, W1, W2P / 32);
    k_ln<<<B, 256, 0, stream>>>(uraw, W1P, nullptr, 0, up1_g, up1_beta, W1, 1,
                                uf, W1P, 1, nullptr, 0);
    k_ln<<<B, 256, 0, stream>>>(uf, W1P, d0f, W1P, skip1_g, skip1_b, W1, 0,
                                nullptr, 0, 1, u216, W1P);
    gemm(W0 / 16, 8, u216, W1P, u0T, W1P, up0_b, u3raw, W0, nullptr, 0, W0, W1P / 32);
    k_ln<<<B, 256, 0, stream>>>(u3raw, W0, nullptr, 0, up0_g, up0_beta, W0, 1,
                                u3f, W0, 1, nullptr, 0);
    // state = ln(u + x0) written straight into the history ring at slot t%M
    int slot = t & (M - 1);
    k_ln<<<B, 256, 0, stream>>>(u3f, W0, x0f, W0, skip0_g, skip0_b, N, 0,
                                ring + slot, N * M, M, nullptr, 0);
    // zp = nlm(hist), logical start = (t+1)%M; f16 copy refreshes concat cols
    k_nlm<<<cdiv(B * N, 256), 256, 0, stream>>>(ring, (t + 1) & (M - 1), nlm_w1,
                                                nlm_b1, nlm_temp, zp, catbuf + D, CAT);
    // output sync -> logits[t]
    k_sync<<<cdiv(B * KO, 256), 256, 0, stream>>>(zp, out_li, out_ri, decay_out,
                                                  ao, bo, syncO16, KO);
    gemm(CP / 16, 8, syncO16, KO, clsT, KO, cls_b, outp + (size_t)t * B * C, C,
         nullptr, 0, C, KO / 32);
  }
}
